// LoRALayer_45337674777318
// MI455X (gfx1250) — compile-verified, matrix operand order checked by reading, not run on hardware
//
#include <hip/hip_runtime.h>

// ---------------------------------------------------------------------------
// LoRA fused layer for gfx1250 (MI455X):
//   out = x @ W^T + (x@A) @ B + bias
// x: (M=16384, 4096) f32, W: (4096, 4096) f32, A: (4096,16), B: (16,4096)
// Compute-bound (AI ~ 950 FLOP/B) -> big GEMM on bf16 WMMA with hi/lo
// f32-emulation (3x v_wmma_f32_16x16x32_bf16 per tile-K, f32 accumulate).
// Double-buffered LDS: one barrier/K-step, global loads issued before the
// WMMA block, conversion VALU placed after it to co-execute with XDL WMMA.
// ---------------------------------------------------------------------------

typedef __attribute__((ext_vector_type(16))) __bf16 bf16x16;
typedef __attribute__((ext_vector_type(8)))  __bf16 bf16x8;
typedef __attribute__((ext_vector_type(4)))  __bf16 bf16x4;
typedef __attribute__((ext_vector_type(8)))  float  f32x8;
typedef __attribute__((ext_vector_type(4)))  float  f32x4;

#define IN_DIM  4096
#define OUT_DIM 4096
#define RANK    16

#define BM 128   // workgroup tile M
#define BN 128   // workgroup tile N
#define BK 32    // K step (one bf16 WMMA K)

// ======================= kernel 1: T = x @ A  (M x 16) =====================
// Memory-bound single pass over x. One wave per row, A staged in LDS chunks,
// wave32 butterfly reduction.
__global__ __launch_bounds__(256) void lora_xa_kernel(
    const float* __restrict__ x, const float* __restrict__ A,
    float* __restrict__ T) {
  __shared__ float sA[128 * RANK];            // 128 K-rows of A (8 KB)
  const int lane = threadIdx.x & 31;
  const int wave = threadIdx.x >> 5;
  const int row  = blockIdx.x * 8 + wave;

  float acc[RANK];
#pragma unroll
  for (int r = 0; r < RANK; ++r) acc[r] = 0.0f;

  const float* xrow = x + (size_t)row * IN_DIM;

  for (int k0 = 0; k0 < IN_DIM; k0 += 128) {
    __syncthreads();
#pragma unroll
    for (int i = 0; i < 8; ++i) {
      int idx = threadIdx.x + i * 256;
      sA[idx] = A[(size_t)k0 * RANK + idx];
    }
    __syncthreads();

    const f32x4 xv = *(const f32x4*)(xrow + k0 + lane * 4);
#pragma unroll
    for (int j = 0; j < 4; ++j) {
      const float* arow = &sA[(lane * 4 + j) * RANK];
#pragma unroll
      for (int r = 0; r < RANK; ++r) acc[r] += xv[j] * arow[r];
    }
  }

  // wave32 butterfly reduction (warpSize == 32 on gfx1250)
#pragma unroll
  for (int r = 0; r < RANK; ++r) {
#pragma unroll
    for (int off = 16; off > 0; off >>= 1)
      acc[r] += __shfl_xor(acc[r], off, 32);
  }
  if (lane < RANK) T[(size_t)row * RANK + lane] = acc[lane];
}

// ============== kernel 2: out = x@W^T (bf16x3 WMMA) + T@B + bias ===========

// Rebuild a 16x32 bf16 A/B fragment from a row-major LDS plane (row stride BK)
// per the CDNA5 16-bit operand layout:
//   lane holds row (lane%16); VGPR0-3 = K[half*8 .. half*8+8),
//   VGPR4-7 = K[16+half*8 .. 16+half*8+8)   (half = lane/16)
__device__ __forceinline__ bf16x16 load_frag(const __bf16* base, int row,
                                             int laneHalf) {
  const bf16x8 a = *(const bf16x8*)(base + row * BK + laneHalf * 8);
  const bf16x8 b = *(const bf16x8*)(base + row * BK + 16 + laneHalf * 8);
  return __builtin_shufflevector(a, b, 0, 1, 2, 3, 4, 5, 6, 7, 8, 9, 10, 11,
                                 12, 13, 14, 15);
}

// f32x4 -> bf16 hi/lo split, store both 8-byte packets to LDS
__device__ __forceinline__ void split_store(__bf16* dh, __bf16* dl, int off,
                                            f32x4 v) {
  bf16x4 h4, l4;
#pragma unroll
  for (int j = 0; j < 4; ++j) {
    __bf16 h = (__bf16)v[j];
    h4[j]    = h;
    l4[j]    = (__bf16)(v[j] - (float)h);
  }
  *(bf16x4*)(dh + off) = h4;
  *(bf16x4*)(dl + off) = l4;
}

__global__ __launch_bounds__(256) void lora_gemm_kernel(
    const float* __restrict__ x,     // M x IN_DIM
    const float* __restrict__ W,     // OUT_DIM x IN_DIM
    const float* __restrict__ T,     // M x RANK  (x@A, from kernel 1)
    const float* __restrict__ Blo,   // RANK x OUT_DIM
    const float* __restrict__ bias,  // OUT_DIM
    float* __restrict__ out) {       // M x OUT_DIM
  // double-buffered bf16 hi/lo planes (4 planes x 8 KB x 2 buffers = 64 KB)
  __shared__ __bf16 sxh[2][BM * BK];
  __shared__ __bf16 sxl[2][BM * BK];
  __shared__ __bf16 swh[2][BN * BK];
  __shared__ __bf16 swl[2][BN * BK];
  __shared__ float  sT[BM * RANK];   // LoRA T tile (8 KB)
  __shared__ float  sB[RANK * BN];   // LoRA B tile (8 KB)
  __shared__ float  sbias[BN];

  const int tid      = threadIdx.x;
  const int lane     = tid & 31;
  const int wave     = tid >> 5;       // 8 waves
  const int laneHalf = lane >> 4;
  const int l16      = lane & 15;

  const int mBase = blockIdx.y * BM;
  const int nBase = blockIdx.x * BN;

  const int wm = (wave & 1) * 64;      // wave sub-tile: 64 (M) x 32 (N)
  const int wn = (wave >> 1) * 32;

  // ---- stage LoRA epilogue operands once (independent LDS regions) ----
#pragma unroll
  for (int i = 0; i < 8; ++i) {
    int idx = tid + i * 256;                          // 0..2047
    sT[idx] = T[(size_t)mBase * RANK + idx];          // rows mBase..+127
    sB[(idx >> 7) * BN + (idx & 127)] =
        Blo[(size_t)(idx >> 7) * OUT_DIM + nBase + (idx & 127)];
  }
  if (tid < BN) sbias[tid] = bias[nBase + tid];

  // per-thread tile-load geometry: 4 float4 per matrix per K-step
  int lrow[4], loff[4];
#pragma unroll
  for (int i = 0; i < 4; ++i) {
    int f   = tid + i * 256;           // float4 index, 0..1023
    lrow[i] = f >> 3;                  // 0..127
    loff[i] = (lrow[i] << 5) + ((f & 7) << 2);  // row*BK + kq
  }

  // ---- prologue: stage K-tile 0 into buffer 0 ----
#pragma unroll
  for (int i = 0; i < 4; ++i) {
    const int kq = loff[i] - (lrow[i] << 5);
    f32x4 xv = *(const f32x4*)(x + (size_t)(mBase + lrow[i]) * IN_DIM + kq);
    f32x4 wv = *(const f32x4*)(W + (size_t)(nBase + lrow[i]) * IN_DIM + kq);
    split_store(sxh[0], sxl[0], loff[i], xv);
    split_store(swh[0], swl[0], loff[i], wv);
  }
  __syncthreads();

  f32x8 acc[4][2] = {};                // 4 M-tiles x 2 N-tiles of 16x16 f32
  int cur = 0;

  for (int kt = 0; kt < IN_DIM; kt += BK) {
    const bool hasNext = (kt + BK) < IN_DIM;

    // ---- issue next tile's global loads before the WMMA block ----
    f32x4 gx[4], gw[4];
    if (hasNext) {
#pragma unroll
      for (int i = 0; i < 4; ++i) {
        const int kq = loff[i] - (lrow[i] << 5);
        gx[i] = *(const f32x4*)(x + (size_t)(mBase + lrow[i]) * IN_DIM + kt +
                                BK + kq);
        gw[i] = *(const f32x4*)(W + (size_t)(nBase + lrow[i]) * IN_DIM + kt +
                                BK + kq);
      }
      if (kt + 2 * BK < IN_DIM) {      // global_prefetch_b8 two tiles ahead
        int prow = tid & 127;
        const float* p =
            (tid < 128)
                ? x + (size_t)(mBase + prow) * IN_DIM + kt + 2 * BK
                : W + (size_t)(nBase + prow) * IN_DIM + kt + 2 * BK;
        __builtin_prefetch(p, 0, 1);
      }
    }

    // ---- fragments from current buffer ----
    const __bf16* pxh = sxh[cur];
    const __bf16* pxl = sxl[cur];
    const __bf16* pwh = swh[cur];
    const __bf16* pwl = swl[cur];
    bf16x16 ah[4], al[4], bh[2], bl[2];
#pragma unroll
    for (int mi = 0; mi < 4; ++mi) {
      int r  = wm + mi * 16 + l16;
      ah[mi] = load_frag(pxh, r, laneHalf);
      al[mi] = load_frag(pxl, r, laneHalf);
    }
#pragma unroll
    for (int ni = 0; ni < 2; ++ni) {
      int r  = wn + ni * 16 + l16;     // column n of B == row n of W tile
      bh[ni] = load_frag(pwh, r, laneHalf);
      bl[ni] = load_frag(pwl, r, laneHalf);
    }

    // ---- f32 emulation: hi*hi + hi*lo + lo*hi, f32 accumulate ----
#pragma unroll
    for (int mi = 0; mi < 4; ++mi) {
#pragma unroll
      for (int ni = 0; ni < 2; ++ni) {
        acc[mi][ni] = __builtin_amdgcn_wmma_f32_16x16x32_bf16(
            false, ah[mi], false, bh[ni], (short)0, acc[mi][ni], false, false);
        acc[mi][ni] = __builtin_amdgcn_wmma_f32_16x16x32_bf16(
            false, ah[mi], false, bl[ni], (short)0, acc[mi][ni], false, false);
        acc[mi][ni] = __builtin_amdgcn_wmma_f32_16x16x32_bf16(
            false, al[mi], false, bh[ni], (short)0, acc[mi][ni], false, false);
      }
    }

    // ---- convert + store next tile into the other buffer (co-executes
    //      as independent VALU while the XDL WMMAs drain) ----
    if (hasNext) {
      const int nxt = cur ^ 1;
#pragma unroll
      for (int i = 0; i < 4; ++i) {
        split_store(sxh[nxt], sxl[nxt], loff[i], gx[i]);
        split_store(swh[nxt], swl[nxt], loff[i], gw[i]);
      }
    }
    __syncthreads();                   // single barrier per K-step
    cur ^= 1;
  }

  // ---- epilogue: + (x@A)@B (exact f32, rank 16 from LDS) + bias, store ----
  // C/D layout: VGPR vr holds M = vr + 8*(lane>=16), N = lane%16.
#pragma unroll
  for (int mi = 0; mi < 4; ++mi) {
#pragma unroll
    for (int ni = 0; ni < 2; ++ni) {
      const int nLoc  = wn + ni * 16 + l16;
      const int nGlob = nBase + nLoc;
      const float bv  = sbias[nLoc];
#pragma unroll
      for (int vr = 0; vr < 8; ++vr) {
        const int mLoc = wm + mi * 16 + vr + 8 * laneHalf;
        float lora = 0.0f;
#pragma unroll
        for (int r = 0; r < RANK; ++r)
          lora += sT[mLoc * RANK + r] * sB[r * BN + nLoc];
        out[(size_t)(mBase + mLoc) * OUT_DIM + nGlob] =
            acc[mi][ni][vr] + lora + bv;
      }
    }
  }
}

// ===========================================================================
extern "C" void kernel_launch(void* const* d_in, const int* in_sizes, int n_in,
                              void* d_out, int out_size, void* d_ws,
                              size_t ws_size, hipStream_t stream) {
  // setup_inputs() order: x, A, B, weight, bias (all f32)
  const float* x    = (const float*)d_in[0];
  const float* A    = (const float*)d_in[1];
  const float* B    = (const float*)d_in[2];
  const float* W    = (const float*)d_in[3];
  const float* bias = (const float*)d_in[4];
  float* out        = (float*)d_out;
  float* T          = (float*)d_ws;  // M x RANK scratch (1 MB)

  const int M = in_sizes[0] / IN_DIM;  // 8*2048 = 16384

  lora_xa_kernel<<<M / 8, 256, 0, stream>>>(x, A, T);

  dim3 grid(OUT_DIM / BN, M / BM);  // 32 x 128 workgroups
  lora_gemm_kernel<<<grid, 256, 0, stream>>>(x, W, T, B, bias, out);
}
